// GTModel_51857435132331
// MI455X (gfx1250) — matrix operand order repaired and checked by reading, not compiled, not run on hardware
//
#include <hip/hip_runtime.h>
#include <hip/hip_bf16.h>

#define N_NODES 30000
#define N_EDGES 480000
#define IN_DIM  1000
#define HID     256
#define HEADS   8
#define DHEAD   32
#define CLS     20
#define HIDP    128
#define BN_EPS  1e-5f

typedef __bf16 v16bf __attribute__((ext_vector_type(16)));
typedef __bf16 v2bf  __attribute__((ext_vector_type(2)));
typedef float  v8f   __attribute__((ext_vector_type(8)));

// ---------- helpers ----------
// native hardware f32->bf16 converts (RTNE); pairs may fuse to packed cvt
__device__ __forceinline__ unsigned pack2_bf16(float x, float y) {
    v2bf p;
    p.x = (__bf16)x;
    p.y = (__bf16)y;
    return __builtin_bit_cast(unsigned, p);
}
__device__ __forceinline__ unsigned short bf16_bits(float x) {
    __bf16 b = (__bf16)x;
    return __builtin_bit_cast(unsigned short, b);
}
// monotonic float<->uint encoding so unsigned atomicMax == float max
__device__ __forceinline__ unsigned flipf(float f) {
    unsigned u = __float_as_uint(f);
    unsigned mask = (unsigned)(-(int)(u >> 31)) | 0x80000000u;
    return u ^ mask;
}
__device__ __forceinline__ float unflipf(unsigned u) {
    unsigned mask = ((u >> 31) - 1u) | 0x80000000u;
    return __uint_as_float(u ^ mask);
}

// ---------- WMMA GEMM: C = epilogue(A[MxK] @ B[KxN]) ----------
// Block tile 128x64, 256 threads = 8 waves in a 4x2 wave grid.
// Each wave computes 32x32 output = 4 x v_wmma_f32_16x16x32_bf16 per 32-wide
// K step (2 A-frags x 2 B-frags, each reused twice).
// LDS tiles are FRAGMENT-MAJOR [subtile][lane][elem]: a fragment is 16
// contiguous bf16 (32B) per lane -> two ds_load_b128.
// Loaders are BRANCH-FREE: M/N overruns are address-clamped (garbage only
// feeds never-stored outputs); only K overrun is zeroed via cndmask-select.
// Epilogue stages C through LDS (aliasing the dead A/B tiles) so global
// writeback is coalesced float4 with one cheap guard per 128-bit store.
// Requires K % 4 == 0 and N % 4 == 0 (true here: K in {1000,256,128},
// N in {256,128,20}).
union GemmSmem {
    struct { unsigned short A[8][32][16]; unsigned short B[4][32][16]; } t;
    float c[128][64];
};

template<bool BIAS, bool RESID, bool RELU>
__global__ __launch_bounds__(256)
void gemm_wmma_bf16(const float* __restrict__ A, const float* __restrict__ B,
                    const float* __restrict__ bias, const float* __restrict__ res,
                    float* __restrict__ C, int M, int K, int N, float scale)
{
    __shared__ __align__(32) GemmSmem sm;

    const int tid  = threadIdx.x;
    const int lane = tid & 31;
    const int wave = tid >> 5;
    const int wm   = wave >> 1;     // 0..3 : 32-row band
    const int wn   = wave & 1;      // 0..1 : 32-col band
    const int half = lane >> 4;
    const int l16  = lane & 15;

    const int blockM = blockIdx.y * 128;
    const int blockN = blockIdx.x * 64;

    v8f acc[2][2] = {};   // [mi][ni]

    for (int kk = 0; kk < K; kk += 32) {
        // ---- A tile: 128x32 f32 -> bf16, fragment-major scatter ----
        #pragma unroll
        for (int p = 0; p < 4; ++p) {
            int r   = p * 32 + (tid >> 3);    // 0..127
            int c   = (tid & 7) * 4;          // 0,4,..,28 (one k-group of 8)
            int gm  = blockM + r;  gm  = gm  < M     ? gm  : M - 1;   // clamp
            int gk0 = kk + c;
            int gkc = gk0 <= K - 4 ? gk0 : K - 4;                     // clamp
            const float4 t = *(const float4*)(A + (size_t)gm * K + gkc);
            const bool ok = gk0 < K;          // whole group in/out (K%4==0)
            float vx = ok ? t.x : 0.f, vy = ok ? t.y : 0.f;
            float vz = ok ? t.z : 0.f, vw = ok ? t.w : 0.f;
            int mSub  = r >> 4, ml = r & 15;
            int g     = c >> 3;                      // 0..3
            int dlane = ((g & 1) << 4) | ml;         // half*16 + mrow
            int i0    = (c & 7) + 8 * (g >> 1);      // elem base (mult of 4)
            *(uint2*)&sm.t.A[mSub][dlane][i0] =
                make_uint2(pack2_bf16(vx, vy), pack2_bf16(vz, vw));  // ds_store_b64
        }

        // ---- B tile: 32x64 f32 -> bf16, fragment-major scatter ----
        #pragma unroll
        for (int p = 0; p < 2; ++p) {
            int r   = p * 16 + (tid >> 4);    // k row 0..31
            int c   = (tid & 15) * 4;         // n 0..60
            int gk  = kk + r;
            int gkc = gk < K      ? gk : K - 1;                      // clamp
            int gn  = blockN + c;
            int gnc = gn <= N - 4 ? gn : N - 4;                      // clamp
            const float4 t = *(const float4*)(B + (size_t)gkc * N + gnc);
            const bool ok = gk < K;
            float vx = ok ? t.x : 0.f, vy = ok ? t.y : 0.f;
            float vz = ok ? t.z : 0.f, vw = ok ? t.w : 0.f;
            int nSub = c >> 4;                 // 0..3
            int ln   = c & 15;                 // base lane-in-half (mult of 4)
            int fh   = (r >> 3) & 1;           // half
            int i    = (r & 7) + 8 * (r >> 4); // elem
            sm.t.B[nSub][(fh << 4) | (ln + 0)][i] = bf16_bits(vx);
            sm.t.B[nSub][(fh << 4) | (ln + 1)][i] = bf16_bits(vy);
            sm.t.B[nSub][(fh << 4) | (ln + 2)][i] = bf16_bits(vz);
            sm.t.B[nSub][(fh << 4) | (ln + 3)][i] = bf16_bits(vw);
        }

        // unconditional clamped prefetch of next A tile (global_prefetch_b8)
        {
            int r  = tid >> 1;
            int gm = blockM + r;          gm = gm < M ? gm : M - 1;
            int gk = kk + 32;             gk = gk < K ? gk : K - 1;
            __builtin_prefetch(&A[(size_t)gm * K + gk], 0, 1);
        }
        __syncthreads();

        // ---- fragment loads: two ds_load_b128 each ----
        union F { v16bf v; uint4 q[2]; };
        F a0, a1, b0, b1;
        {
            const uint4* pa0 = (const uint4*)sm.t.A[2 * wm + 0][lane];
            const uint4* pa1 = (const uint4*)sm.t.A[2 * wm + 1][lane];
            const uint4* pb0 = (const uint4*)sm.t.B[2 * wn + 0][lane];
            const uint4* pb1 = (const uint4*)sm.t.B[2 * wn + 1][lane];
            a0.q[0] = pa0[0]; a0.q[1] = pa0[1];
            a1.q[0] = pa1[0]; a1.q[1] = pa1[1];
            b0.q[0] = pb0[0]; b0.q[1] = pb0[1];
            b1.q[0] = pb1[0]; b1.q[1] = pb1[1];
        }

        acc[0][0] = __builtin_amdgcn_wmma_f32_16x16x32_bf16(
                        false, a0.v, false, b0.v, (short)0, acc[0][0], false, false);
        acc[0][1] = __builtin_amdgcn_wmma_f32_16x16x32_bf16(
                        false, a0.v, false, b1.v, (short)0, acc[0][1], false, false);
        acc[1][0] = __builtin_amdgcn_wmma_f32_16x16x32_bf16(
                        false, a1.v, false, b0.v, (short)0, acc[1][0], false, false);
        acc[1][1] = __builtin_amdgcn_wmma_f32_16x16x32_bf16(
                        false, a1.v, false, b1.v, (short)0, acc[1][1], false, false);
        __syncthreads();   // also fences tiles before the epilogue aliases them
    }

    // ---- epilogue stage 1: accumulators -> LDS (row-major, unguarded) ----
    // C/D layout: VGPR r holds M = r + 8*half, N = l16
    #pragma unroll
    for (int mi = 0; mi < 2; ++mi)
        #pragma unroll
        for (int ni = 0; ni < 2; ++ni)
            #pragma unroll
            for (int r = 0; r < 8; ++r)
                sm.c[wm * 32 + mi * 16 + half * 8 + r][wn * 32 + ni * 16 + l16]
                    = acc[mi][ni][r];
    __syncthreads();

    // ---- epilogue stage 2: coalesced float4 writeback with fused math ----
    #pragma unroll
    for (int p = 0; p < 8; ++p) {
        int row = p * 16 + (tid >> 4);     // 0..127
        int col = (tid & 15) * 4;          // 0..60
        int gm  = blockM + row;
        int gn  = blockN + col;
        if (gm < M && gn + 4 <= N) {
            float4 v = *(const float4*)&sm.c[row][col];     // ds_load_b128
            if (BIAS) {
                const float4 bb = *(const float4*)(bias + gn);
                v.x += bb.x; v.y += bb.y; v.z += bb.z; v.w += bb.w;
            }
            v.x *= scale; v.y *= scale; v.z *= scale; v.w *= scale;
            if (RESID) {
                const float4 rr = *(const float4*)(res + (size_t)gm * N + gn);
                v.x += rr.x; v.y += rr.y; v.z += rr.z; v.w += rr.w;
            }
            if (RELU) {
                v.x = fmaxf(v.x, 0.f); v.y = fmaxf(v.y, 0.f);
                v.z = fmaxf(v.z, 0.f); v.w = fmaxf(v.w, 0.f);
            }
            *(float4*)(C + (size_t)gm * N + gn) = v;        // global_store_b128
        }
    }
}

// ---------- edge-sparse attention ----------
// q/k/v logical layout: [N, DHEAD, HEADS] flattened as x[i, d*HEADS + h]
__global__ void edge_scores_kernel(const int* __restrict__ rows, const int* __restrict__ cols,
                                   const float* __restrict__ q, const float* __restrict__ k,
                                   float* __restrict__ s, unsigned* __restrict__ mEnc)
{
    int idx = blockIdx.x * blockDim.x + threadIdx.x;
    if (idx >= N_EDGES * HEADS) return;
    int e = idx >> 3, h = idx & 7;
    int r = rows[e], c = cols[e];
    const float* qp = q + (size_t)r * HID + h;
    const float* kp = k + (size_t)c * HID + h;
    float acc = 0.f;
    #pragma unroll
    for (int d = 0; d < DHEAD; ++d)
        acc = fmaf(qp[d * HEADS], kp[d * HEADS], acc);
    s[idx] = acc;
    atomicMax(&mEnc[r * HEADS + h], flipf(acc));
}

__global__ void edge_expz_kernel(const int* __restrict__ rows, float* __restrict__ s,
                                 const unsigned* __restrict__ mEnc, float* __restrict__ z)
{
    int idx = blockIdx.x * blockDim.x + threadIdx.x;
    if (idx >= N_EDGES * HEADS) return;
    int e = idx >> 3, h = idx & 7;
    int r = rows[e];
    float p = __expf(s[idx] - unflipf(mEnc[r * HEADS + h]));
    s[idx] = p;
    atomicAdd(&z[r * HEADS + h], p);
}

__global__ void edge_aggr_kernel(const int* __restrict__ rows, const int* __restrict__ cols,
                                 const float* __restrict__ s, const float* __restrict__ z,
                                 const float* __restrict__ v, float* __restrict__ out)
{
    int idx = blockIdx.x * blockDim.x + threadIdx.x;
    if (idx >= N_EDGES * HEADS) return;
    int e = idx >> 3, h = idx & 7;
    int r = rows[e], c = cols[e];
    float a = s[idx] / z[r * HEADS + h];
    const float* vp = v + (size_t)c * HID + h;
    float* op = out + (size_t)r * HID + h;
    #pragma unroll
    for (int d = 0; d < DHEAD; ++d)
        atomicAdd(&op[d * HEADS], a * vp[d * HEADS]);
}

// ---------- batchnorm (over rows, per column) ----------
__global__ __launch_bounds__(HID)
void bn_partial_kernel(const float* __restrict__ x, float* __restrict__ sum,
                       float* __restrict__ sumsq, int M)
{
    int c  = threadIdx.x;             // column (coalesced across threads)
    int r0 = blockIdx.x * 256;
    float s = 0.f, s2 = 0.f;
    for (int r = 0; r < 256; ++r) {
        int rr = r0 + r;
        if (rr < M) {
            float v = x[(size_t)rr * HID + c];
            s += v; s2 = fmaf(v, v, s2);
        }
    }
    atomicAdd(&sum[c], s);
    atomicAdd(&sumsq[c], s2);
}

__global__ void bn_apply_kernel(float* __restrict__ x, const float* __restrict__ sum,
                                const float* __restrict__ sumsq,
                                const float* __restrict__ g, const float* __restrict__ b, int M)
{
    int idx = blockIdx.x * blockDim.x + threadIdx.x;
    if (idx >= M * HID) return;
    int c = idx & (HID - 1);
    float inv = 1.f / (float)M;
    float mu  = sum[c] * inv;
    float var = sumsq[c] * inv - mu * mu;
    x[idx] = g[c] * (x[idx] - mu) * rsqrtf(var + BN_EPS) + b[c];
}

// ---------- fills ----------
__global__ void fill_f32_kernel(float* p, float v, int n) {
    int i = blockIdx.x * blockDim.x + threadIdx.x;
    if (i < n) p[i] = v;
}
__global__ void fill_u32_kernel(unsigned* p, unsigned v, int n) {
    int i = blockIdx.x * blockDim.x + threadIdx.x;
    if (i < n) p[i] = v;
}

// ---------- host launch ----------
extern "C" void kernel_launch(void* const* d_in, const int* in_sizes, int n_in,
                              void* d_out, int out_size, void* d_ws, size_t ws_size,
                              hipStream_t stream) {
    (void)in_sizes; (void)n_in; (void)out_size; (void)ws_size;

    const int*   edge = (const int*)d_in[0];
    const float* feat = (const float*)d_in[1];
    const float* Wemb = (const float*)d_in[2];
    const float* bemb = (const float*)d_in[3];
    const float *Wq[2], *bq[2], *Wk[2], *bk[2], *Wv[2], *bv[2], *WO[2], *bO[2], *gam[2], *bet[2];
    for (int l = 0; l < 2; ++l) {
        int base = 4 + l * 10;
        Wq[l]  = (const float*)d_in[base + 0]; bq[l]  = (const float*)d_in[base + 1];
        Wk[l]  = (const float*)d_in[base + 2]; bk[l]  = (const float*)d_in[base + 3];
        Wv[l]  = (const float*)d_in[base + 4]; bv[l]  = (const float*)d_in[base + 5];
        WO[l]  = (const float*)d_in[base + 6]; bO[l]  = (const float*)d_in[base + 7];
        gam[l] = (const float*)d_in[base + 8]; bet[l] = (const float*)d_in[base + 9];
    }
    const float* Wp1 = (const float*)d_in[24];
    const float* bp1 = (const float*)d_in[25];
    const float* Wp2 = (const float*)d_in[26];
    const float* bp2 = (const float*)d_in[27];

    const int* rows = edge;             // edge_index[0]
    const int* cols = edge + N_EDGES;   // edge_index[1]

    // workspace layout
    char* ws = (char*)d_ws;
    size_t off = 0;
    auto wsalloc = [&](size_t bytes) -> void* {
        void* p = ws + off;
        off += (bytes + 255) & ~(size_t)255;
        return p;
    };
    const size_t hbytes = sizeof(float) * (size_t)N_NODES * HID;
    float*    h0    = (float*)wsalloc(hbytes);
    float*    h1    = (float*)wsalloc(hbytes);
    float*    qb    = (float*)wsalloc(hbytes);   // reused as attn-out & MLP hidden
    float*    kb    = (float*)wsalloc(hbytes);
    float*    vb    = (float*)wsalloc(hbytes);
    float*    sb    = (float*)wsalloc(sizeof(float) * (size_t)N_EDGES * HEADS);
    unsigned* mEnc  = (unsigned*)wsalloc(sizeof(unsigned) * (size_t)N_NODES * HEADS);
    float*    zb    = (float*)wsalloc(sizeof(float) * (size_t)N_NODES * HEADS);
    float*    bnsum = (float*)wsalloc(sizeof(float) * 2 * HID);
    float*    bnsq  = bnsum + HID;

    const dim3 blk(256);
    auto ggrid = [](int M, int N) { return dim3((unsigned)((N + 63) / 64), (unsigned)((M + 127) / 128)); };
    const int NH  = N_NODES * HEADS;
    const int EH  = N_EDGES * HEADS;
    const int ehB = (EH + 255) / 256;
    const int nhB = (NH + 255) / 256;
    const int elB = (N_NODES * HID + 255) / 256;
    const float scaleQ = 0.17677669529663689f;   // DHEAD^-0.5

    // embedding: h0 = features @ Wemb + bemb
    gemm_wmma_bf16<true, false, false><<<ggrid(N_NODES, HID), blk, 0, stream>>>(
        feat, Wemb, bemb, nullptr, h0, N_NODES, IN_DIM, HID, 1.f);

    float* hin = h0;
    float* hout = h1;
    for (int l = 0; l < 2; ++l) {
        // q/k/v projections (q scaled by d_head^-0.5 after bias, per reference)
        gemm_wmma_bf16<true, false, false><<<ggrid(N_NODES, HID), blk, 0, stream>>>(
            hin, Wq[l], bq[l], nullptr, qb, N_NODES, HID, HID, scaleQ);
        gemm_wmma_bf16<true, false, false><<<ggrid(N_NODES, HID), blk, 0, stream>>>(
            hin, Wk[l], bk[l], nullptr, kb, N_NODES, HID, HID, 1.f);
        gemm_wmma_bf16<true, false, false><<<ggrid(N_NODES, HID), blk, 0, stream>>>(
            hin, Wv[l], bv[l], nullptr, vb, N_NODES, HID, HID, 1.f);

        // sparse softmax-attention over edges
        fill_u32_kernel<<<nhB, blk, 0, stream>>>(mEnc, 0u, NH);   // 0 == min flipped float
        fill_f32_kernel<<<nhB, blk, 0, stream>>>(zb, 0.f, NH);
        edge_scores_kernel<<<ehB, blk, 0, stream>>>(rows, cols, qb, kb, sb, mEnc);
        edge_expz_kernel<<<ehB, blk, 0, stream>>>(rows, sb, mEnc, zb);
        fill_f32_kernel<<<elB, blk, 0, stream>>>(qb, 0.f, N_NODES * HID);  // reuse q as attn-out
        edge_aggr_kernel<<<ehB, blk, 0, stream>>>(rows, cols, sb, zb, vb, qb);

        // output projection + residual
        gemm_wmma_bf16<true, true, false><<<ggrid(N_NODES, HID), blk, 0, stream>>>(
            qb, WO[l], bO[l], hin, hout, N_NODES, HID, HID, 1.f);

        // batchnorm (in place on hout)
        fill_f32_kernel<<<(2 * HID + 255) / 256, blk, 0, stream>>>(bnsum, 0.f, 2 * HID);
        bn_partial_kernel<<<(N_NODES + 255) / 256, dim3(HID), 0, stream>>>(hout, bnsum, bnsq, N_NODES);
        bn_apply_kernel<<<elB, blk, 0, stream>>>(hout, bnsum, bnsq, gam[l], bet[l], N_NODES);

        float* t = hin; hin = hout; hout = t;
    }

    // MLP head: relu(h @ Wp1 + bp1) @ Wp2 + bp2  -> d_out [N, CLS]
    float* hidden = qb;  // reuse
    gemm_wmma_bf16<true, false, true><<<ggrid(N_NODES, HIDP), blk, 0, stream>>>(
        hin, Wp1, bp1, nullptr, hidden, N_NODES, HID, HIDP, 1.f);
    gemm_wmma_bf16<true, false, false><<<ggrid(N_NODES, CLS), blk, 0, stream>>>(
        hidden, Wp2, bp2, nullptr, (float*)d_out, N_NODES, HIDP, CLS, 1.f);
}
